// Unet_25271587570108
// MI455X (gfx1250) — matrix operand order repaired and checked by reading, not compiled, not run on hardware
//
#include <hip/hip_runtime.h>
#include <hip/hip_bf16.h>
#include <math.h>

// CDNA5 wave32 WMMA fragment types
typedef __attribute__((ext_vector_type(16))) __bf16 v16bf;
typedef __attribute__((ext_vector_type(8)))  float  v8f;

union Frag { uint32_t u[8]; v16bf v; };   // u[i] holds bf16 elements 2i, 2i+1

#define B_     128
#define NN     128                       // padded Hadamard size
#define IMGPX  (NN * NN)                 // 16384
#define NIMG   (B_ * 4)                  // 512 images of 128x128
#define PLANE  ((size_t)NIMG * IMGPX)    // 8,388,608 floats per [B,4,128,128] buffer
#define H1_    122
#define PITCH  66                        // LDS row pitch in dwords (132 bf16): bank step 2

__device__ __forceinline__ float sgm(float a) { return 1.0f / (1.0f + __expf(-a)); }

__device__ __forceinline__ uint32_t pack_bf16(float a, float b) {
    const __bf16 ha = (__bf16)a, hb = (__bf16)b;
    return ((uint32_t)__builtin_bit_cast(unsigned short, hb) << 16) |
            (uint32_t)__builtin_bit_cast(unsigned short, ha);
}
// packed (+/-1, +/-1) bf16 pair for Hadamard entries at (r&k), (r&(k+1)), k even
__device__ __forceinline__ uint32_t hpair(int k_even, int r, uint32_t rlow) {
    const uint32_t p0 = (uint32_t)(__builtin_popcount(k_even & r) & 1);
    return 0x3F803F80u | (p0 << 15) | ((p0 ^ rlow) << 31);
}

// ---------------------------------------------------------------------------
// T = X * H per 128x128 image. One block = one tile-row (16 rows x 128 cols):
// the 16x128 f32 A-slab is staged to LDS as bf16 hi/lo ONCE and shared by all
// 8 waves (8 tn-tiles). H (+/-1, exact bf16) is synthesized in-register.
// Dual accumulators keep the two WMMA streams independent.
// ---------------------------------------------------------------------------
__global__ void had_rowT(const float* __restrict__ in, float* __restrict__ out)
{
    __shared__ uint32_t AhiU[16 * PITCH];
    __shared__ uint32_t AloU[16 * PITCH];
    const int tid  = threadIdx.x;
    const int img  = blockIdx.x >> 3;
    const int tm0  = (blockIdx.x & 7) << 4;
    const float* X = in  + (size_t)img * IMGPX + (size_t)tm0 * NN;
    float*       T = out + (size_t)img * IMGPX;

    // ---- stage 16x128 slab: coalesced float4 loads, split to bf16 hi/lo ----
    {
        const int row  = tid >> 4;            // 0..15
        const int col0 = (tid & 15) * 8;      // 0,8,...,120
        const float4 f0 = *(const float4*)(X + row * NN + col0);
        const float4 f1 = *(const float4*)(X + row * NN + col0 + 4);
        const float f[8] = { f0.x, f0.y, f0.z, f0.w, f1.x, f1.y, f1.z, f1.w };
        const int base = row * PITCH + (col0 >> 1);
#pragma unroll
        for (int j = 0; j < 4; ++j) {
            const __bf16 h0 = (__bf16)f[2 * j], h1 = (__bf16)f[2 * j + 1];
            AhiU[base + j] = pack_bf16(f[2 * j], f[2 * j + 1]);
            AloU[base + j] = pack_bf16(f[2 * j] - (float)h0, f[2 * j + 1] - (float)h1);
        }
        __builtin_prefetch(X + 16 * NN + row * NN + col0, 0, 1);  // next slab
    }
    __syncthreads();

    const int lane  = tid & 31;
    const int halfL = lane >> 4;
    const int l15   = lane & 15;
    const int tn    = (tid >> 5) << 4;
    const int ncol  = tn + l15;
    const uint32_t nlow = (uint32_t)(ncol & 1);

    v8f accH = {}, accL = {};
    for (int k0 = 0; k0 < NN; k0 += 32) {
        Frag ahi, alo, bh;
        const int rbase = l15 * PITCH;
#pragma unroll
        for (int v = 0; v < 8; ++v) {
            // A (16x32) layout: lanes0-15 K=0..7,16..23; lanes16-31 K=8..15,24..31
            const int kau = (k0 >> 1) + (v & 3) + (halfL << 2) + ((v >> 2) << 3);
            ahi.u[v] = AhiU[rbase + kau];
            alo.u[v] = AloU[rbase + kau];
            // B (32x16) layout: lanes0-15 K=0..7,16..23; lanes16-31 K=8..15,24..31
            const int kb = k0 + ((v & 3) * 2) + (halfL * 8) + ((v >> 2) << 4);
            bh.u[v] = hpair(kb, ncol, nlow);
        }
        accH = __builtin_amdgcn_wmma_f32_16x16x32_bf16(false, ahi.v, false, bh.v, (short)0, accH, false, false);
        accL = __builtin_amdgcn_wmma_f32_16x16x32_bf16(false, alo.v, false, bh.v, (short)0, accL, false, false);
    }
#pragma unroll
    for (int r = 0; r < 8; ++r)
        T[(tm0 + r + halfL * 8) * NN + ncol] = accH[r] + accL[r];
}

// ---------------------------------------------------------------------------
// Y = H * T, fused epilogue. One block = one 16-column stripe: the 128x16 f32
// B-slab is staged (transposed) to LDS as bf16 hi/lo once; 8 waves cover the
// 8 tm-tiles of the stripe.
//   doThresh=1: q = v*Y; out = (|q|>|T|? q : 0) (+ addSrc); KL sigmoid buckets
//   doThresh=0: out = Y * scale  (folds the two 1/128 of ifwht2)
// ---------------------------------------------------------------------------
__global__ void had_colT(const float* __restrict__ Tin, float* __restrict__ out,
                         const float* __restrict__ vmat, const float* __restrict__ tmat,
                         const float* __restrict__ addSrc, float* __restrict__ klAcc,
                         float scale, int doThresh)
{
    __shared__ uint32_t BhiU[16 * PITCH];  // [col][k] transposed slab
    __shared__ uint32_t BloU[16 * PITCH];
    __shared__ float klb[64];
    const int tid = threadIdx.x;
    if (tid < 64) klb[tid] = 0.0f;

    const int img = blockIdx.x >> 3;
    const int tn0 = (blockIdx.x & 7) << 4;
    const float* T = Tin + (size_t)img * IMGPX;
    float*       O = out + (size_t)img * IMGPX;

    // ---- stage 128x16 slab transposed: rows coalesced, store by column ----
    {
        const int colS = tid & 15;
        const int krow = (tid >> 4) * 2;    // 0,2,...,30
#pragma unroll
        for (int pass = 0; pass < 4; ++pass) {
            const int k = pass * 32 + krow;
            const float b0 = T[k * NN + tn0 + colS];
            const float b1 = T[(k + 1) * NN + tn0 + colS];
            const __bf16 h0 = (__bf16)b0, h1 = (__bf16)b1;
            const int idx = colS * PITCH + (k >> 1);
            BhiU[idx] = pack_bf16(b0, b1);
            BloU[idx] = pack_bf16(b0 - (float)h0, b1 - (float)h1);
        }
        __builtin_prefetch(T + IMGPX + (krow)*NN + tn0 + colS, 0, 1);  // next image
    }
    __syncthreads();

    const int lane  = tid & 31;
    const int halfL = lane >> 4;
    const int l15   = lane & 15;
    const int tm    = (tid >> 5) << 4;
    const int mrow  = tm + l15;                 // H row (A operand, exact)
    const int ncol  = tn0 + l15;                // output column
    const uint32_t mlow = (uint32_t)(mrow & 1);

    v8f accH = {}, accL = {};
    for (int k0 = 0; k0 < NN; k0 += 32) {
        Frag ah, bhi, blo;
        const int cbase = l15 * PITCH;
#pragma unroll
        for (int v = 0; v < 8; ++v) {
            const int ka = k0 + ((v & 3) * 2) + (halfL * 8) + ((v >> 2) << 4);
            ah.u[v] = hpair(ka, mrow, mlow);
            const int kbu = (k0 >> 1) + (v & 3) + (halfL << 2) + ((v >> 2) << 3);
            bhi.u[v] = BhiU[cbase + kbu];
            blo.u[v] = BloU[cbase + kbu];
        }
        accH = __builtin_amdgcn_wmma_f32_16x16x32_bf16(false, ah.v, false, bhi.v, (short)0, accH, false, false);
        accL = __builtin_amdgcn_wmma_f32_16x16x32_bf16(false, ah.v, false, blo.v, (short)0, accL, false, false);
    }

    float klLocal = 0.0f;
#pragma unroll
    for (int r = 0; r < 8; ++r) {
        const int row    = tm + r + halfL * 8;
        const size_t idx = (size_t)img * IMGPX + row * NN + ncol;
        const float y    = accH[r] + accL[r];
        float o;
        if (doThresh) {
            const float q  = vmat[row * NN + ncol] * y;
            const float aT = fabsf(tmat[row * NN + ncol]);
            const float rr = (fabsf(q) > aT) ? q : 0.0f;   // exact hard-threshold
            klLocal += sgm(fabsf(rr));
            o = rr + (addSrc ? addSrc[idx] : 0.0f);
        } else {
            o = y * scale;
        }
        O[row * NN + ncol] = o;
    }

    if (doThresh) atomicAdd(&klb[ncol & 63], klLocal);     // ds_add_f32
    __syncthreads();
    if (doThresh && tid < 64) atomicAdd(&klAcc[tid], klb[tid]);
}

// ---------------------------------------------------------------------------
// conv1: x[128,1,256,256] * w1[4,1,16,16], stride 2, pad 1 -> [128,4,128,128]
// (zeros outside the 122x122 valid region so the Hadamard pad is built in)
// ---------------------------------------------------------------------------
__global__ void conv1_k(const float* __restrict__ x, const float* __restrict__ w,
                        const float* __restrict__ b, float* __restrict__ out)
{
    __shared__ float wsh[1024];
    const int tid = threadIdx.x;
    for (int i = tid; i < 1024; i += 256) wsh[i] = w[i];
    __syncthreads();
    const int idx = blockIdx.x * 256 + tid;
    const int wc  = idx & 127;
    const int hr  = (idx >> 7) & 127;
    const int c   = (idx >> 14) & 3;
    const int bi  = idx >> 16;
    float acc = 0.0f;
    const bool valid = (hr < H1_) && (wc < H1_);
    if (valid) {
        acc = b[c];
        const float* xp = x + (size_t)bi * 65536;
        const float* wp = wsh + c * 256;
        const int ih0 = hr * 2 - 1, iw0 = wc * 2 - 1;
        for (int kh = 0; kh < 16; ++kh) {
            const int ih = ih0 + kh;
            if ((unsigned)ih < 256u) {
                const float* xr = xp + ih * 256;
                const float* wr = wp + kh * 16;
#pragma unroll
                for (int kw = 0; kw < 16; ++kw) {
                    const int iw = iw0 + kw;
                    if ((unsigned)iw < 256u) acc += xr[iw] * wr[kw];
                }
            }
        }
    }
    out[idx] = valid ? acc : 0.0f;
}

// ---------------------------------------------------------------------------
// 7x7 conv, 4->4 channels, pad 3, valid region 122x122 of 128-strided planes.
// transposed=1 loads wf[o][i][kh][kw] = w[i][o][6-kh][6-kw] (ConvTranspose2d).
// Optional fused residual add (x19 = convT(x18)+x10).
// ---------------------------------------------------------------------------
__global__ void conv7_k(const float* __restrict__ in, const float* __restrict__ w,
                        const float* __restrict__ b, const float* __restrict__ addSrc,
                        float* __restrict__ out, int transposed)
{
    __shared__ float wsh[784];
    const int tid = threadIdx.x;
    for (int i = tid; i < 784; i += 256) {
        if (transposed) {
            const int o  = i / 196;
            const int ci = (i / 49) & 3;
            const int kk = i % 49;
            const int kh = kk / 7, kw = kk % 7;
            wsh[i] = w[ci * 196 + o * 49 + (6 - kh) * 7 + (6 - kw)];
        } else {
            wsh[i] = w[i];
        }
    }
    __syncthreads();
    const int idx = blockIdx.x * 256 + tid;
    const int wc  = idx & 127;
    const int hr  = (idx >> 7) & 127;
    const int c   = (idx >> 14) & 3;
    const int bi  = idx >> 16;
    float acc = 0.0f;
    const bool valid = (hr < H1_) && (wc < H1_);
    if (valid) {
        acc = b[c];
        const float* ip = in + (size_t)bi * 4 * IMGPX;
        for (int ci = 0; ci < 4; ++ci) {
            const float* pl = ip + ci * IMGPX;
            const float* wp = wsh + c * 196 + ci * 49;
#pragma unroll
            for (int kh = 0; kh < 7; ++kh) {
                const int ih = hr - 3 + kh;
                if ((unsigned)ih < (unsigned)H1_) {
                    const float* xr = pl + ih * NN;
#pragma unroll
                    for (int kw = 0; kw < 7; ++kw) {
                        const int iw = wc - 3 + kw;
                        if ((unsigned)iw < (unsigned)H1_) acc += xr[iw] * wp[kh * 7 + kw];
                    }
                }
            }
        }
        if (addSrc) acc += addSrc[idx];
    }
    out[idx] = valid ? acc : 0.0f;
}

// ---------------------------------------------------------------------------
// convT: x27[128,4,122,122] * w4[4,1,16,16], stride 2, pad 1 -> [128,1,256,256]
// fused with final elementwise * x.
// ---------------------------------------------------------------------------
__global__ void conv4t_k(const float* __restrict__ x27, const float* __restrict__ w4,
                         const float* __restrict__ b4, const float* __restrict__ x,
                         float* __restrict__ out)
{
    __shared__ float wsh[1024];
    const int tid = threadIdx.x;
    for (int i = tid; i < 1024; i += 256) wsh[i] = w4[i];
    __syncthreads();
    const int idx = blockIdx.x * 256 + tid;
    const int ow  = idx & 255;
    const int oh  = (idx >> 8) & 255;
    const int bi  = idx >> 16;
    const float* ip = x27 + (size_t)bi * 4 * IMGPX;
    float acc = b4[0];
    const int ph = (oh + 1) & 1;   // kh parity with oh = 2*ih - 1 + kh
    const int pw = (ow + 1) & 1;
#pragma unroll
    for (int kkh = 0; kkh < 8; ++kkh) {
        const int kh = 2 * kkh + ph;
        const int th = oh + 1 - kh;
        const int ih = th >> 1;
        if (th >= 0 && ih < H1_) {
#pragma unroll
            for (int kkw = 0; kkw < 8; ++kkw) {
                const int kw = 2 * kkw + pw;
                const int tw = ow + 1 - kw;
                const int iw = tw >> 1;
                if (tw >= 0 && iw < H1_) {
                    const float* wp = wsh + kh * 16 + kw;
#pragma unroll
                    for (int c = 0; c < 4; ++c)
                        acc += ip[c * IMGPX + ih * NN + iw] * wp[c * 256];
                }
            }
        }
    }
    out[idx] = acc * x[idx];
}

// ---------------------------------------------------------------------------
// TV: sum |dh| + |dw| over [128,1,256,256], block tree-reduce + global atomic
// ---------------------------------------------------------------------------
__global__ void tv_k(const float* __restrict__ o, float* __restrict__ tvAcc)
{
    __shared__ float red[256];
    const int tid = threadIdx.x;
    const int idx = blockIdx.x * 256 + tid;
    const int ww  = idx & 255;
    const int hh  = (idx >> 8) & 255;
    const float v = o[idx];
    float s = 0.0f;
    if (hh < 255) s += fabsf(o[idx + 256] - v);
    if (ww < 255) s += fabsf(o[idx + 1] - v);
    red[tid] = s;
    __syncthreads();
    for (int off = 128; off > 0; off >>= 1) {
        if (tid < off) red[tid] += red[tid + off];
        __syncthreads();
    }
    if (tid == 0) atomicAdd(tvAcc, red[0]);
}

__global__ void init_k(float* __restrict__ scr) { scr[threadIdx.x] = 0.0f; }

__global__ void fin_k(const float* __restrict__ scr, float* __restrict__ lossOut)
{
    if (threadIdx.x == 0) {
        const float p    = 1.0f / (1.0f + expf(-0.001f));
        const float einv = 64.0f / (float)PLANE;   // 1/e, e = size/64
        float tot = 0.0f;
        for (int s = 0; s < 3; ++s)
            for (int j = 0; j < 64; ++j) {
                const float qm = scr[s * 64 + j] * einv;
                tot += p * logf(p / qm) + (1.0f - p) * logf((1.0f - p) / (1.0f - qm));
            }
        const float tvterm = 0.05f * scr[192] / 8388608.0f;
        lossOut[0] = tvterm + 0.1f * tot;
    }
}

extern "C" void kernel_launch(void* const* d_in, const int* in_sizes, int n_in,
                              void* d_out, int out_size, void* d_ws, size_t ws_size,
                              hipStream_t stream)
{
    const float* x  = (const float*)d_in[0];
    const float* v1 = (const float*)d_in[2];
    const float* T1 = (const float*)d_in[3];
    const float* v2 = (const float*)d_in[4];
    const float* T2 = (const float*)d_in[5];
    const float* v3 = (const float*)d_in[6];
    const float* T3 = (const float*)d_in[7];
    const float* w1 = (const float*)d_in[8];
    const float* b1 = (const float*)d_in[9];
    const float* w2 = (const float*)d_in[10];
    const float* b2 = (const float*)d_in[11];
    const float* w3 = (const float*)d_in[12];
    const float* b3 = (const float*)d_in[13];
    const float* w4 = (const float*)d_in[14];
    const float* b4 = (const float*)d_in[15];
    float* out = (float*)d_out;
    float* ws  = (float*)d_ws;

    float* bufA  = ws;               // 32 MB each
    float* bufB  = ws + PLANE;
    float* bufC  = ws + 2 * PLANE;
    float* bufT  = ws + 3 * PLANE;
    float* bufX6 = ws + 4 * PLANE;
    float* scr   = ws + 5 * PLANE;   // 192 KL buckets + 1 TV accumulator

    const dim3 blk(256);
    const dim3 gC(32768);            // 8,388,608 elements / 256
    const dim3 gH(4096);             // 512 imgs * 8 slab-stripes
    const float INV = 1.0f / 16384.0f;

    init_k<<<1, 256, 0, stream>>>(scr);
    conv1_k<<<gC, blk, 0, stream>>>(x, w1, b1, bufA);                         // x1 (padded)

    had_rowT<<<gH, blk, 0, stream>>>(bufA, bufT);                             // fwht2
    had_colT<<<gH, blk, 0, stream>>>(bufT, bufX6, v1, T1, nullptr, scr, 1.0f, 1);      // x6

    had_rowT<<<gH, blk, 0, stream>>>(bufX6, bufT);                            // ifwht2
    had_colT<<<gH, blk, 0, stream>>>(bufT, bufA, nullptr, nullptr, nullptr, nullptr, INV, 0); // x9

    conv7_k<<<gC, blk, 0, stream>>>(bufA, w2, b2, nullptr, bufB, 0);          // x10

    had_rowT<<<gH, blk, 0, stream>>>(bufB, bufT);
    had_colT<<<gH, blk, 0, stream>>>(bufT, bufA, v2, T2, nullptr, scr + 64, 1.0f, 1);  // x15

    had_rowT<<<gH, blk, 0, stream>>>(bufA, bufT);
    had_colT<<<gH, blk, 0, stream>>>(bufT, bufA, nullptr, nullptr, nullptr, nullptr, INV, 0); // x18

    conv7_k<<<gC, blk, 0, stream>>>(bufA, w3, b3, bufB, bufC, 1);             // x19 = convT+x10

    had_rowT<<<gH, blk, 0, stream>>>(bufC, bufT);
    had_colT<<<gH, blk, 0, stream>>>(bufT, bufA, v3, T3, bufX6, scr + 128, 1.0f, 1);   // x24+x6

    had_rowT<<<gH, blk, 0, stream>>>(bufA, bufT);
    had_colT<<<gH, blk, 0, stream>>>(bufT, bufB, nullptr, nullptr, nullptr, nullptr, INV, 0); // x27

    conv4t_k<<<gC, blk, 0, stream>>>(bufB, w4, b4, x, out);                   // out = convT * x

    tv_k<<<gC, blk, 0, stream>>>(out, scr + 192);
    fin_k<<<1, 64, 0, stream>>>(scr, out + (out_size - 1));
}
